// Pseudopod_31035433681162
// MI455X (gfx1250) — matrix operand order repaired and checked by reading, not compiled, not run on hardware
//
#include <hip/hip_runtime.h>

#define SEQ 8192
#define DIM 128
#define BN  64
#define BM  64
#define NWAVES 4

typedef __attribute__((ext_vector_type(16))) __bf16 v16bf;
typedef __attribute__((ext_vector_type(2)))  __bf16 v2bf;
typedef __attribute__((ext_vector_type(8)))  float  v8f;

#ifndef __has_builtin
#define __has_builtin(x) 0
#endif
#if __has_builtin(__builtin_amdgcn_cvt_pk_bf16_f32)
#define HAVE_PK_BF16 1
#else
#define HAVE_PK_BF16 0
#endif

union BF16Frag {
    v16bf v;
    v2bf  p[8];
    unsigned short s[16];
    uint4 q[2];
};

union BF16Pair {
    v2bf v;
    unsigned short s[2];
};

// Native f32 -> bf16 (lets the backend pick the hardware convert; the
// compiler expands with correct RNE semantics if no instruction exists).
__device__ __forceinline__ unsigned short f2bf(float x) {
    return __builtin_bit_cast(unsigned short, (__bf16)x);
}

__device__ __forceinline__ void pack2(BF16Frag& f, int j, float a, float b) {
#if HAVE_PK_BF16
    f.p[j] = __builtin_amdgcn_cvt_pk_bf16_f32(a, b);   // v_cvt_pk_bf16_f32
#else
    v2bf t;
    t[0] = (__bf16)a;
    t[1] = (__bf16)b;
    f.p[j] = t;
#endif
}

__device__ __forceinline__ BF16Pair packpair(float a, float b) {
    BF16Pair r;
#if HAVE_PK_BF16
    r.v = __builtin_amdgcn_cvt_pk_bf16_f32(a, b);
#else
    v2bf t;
    t[0] = (__bf16)a;
    t[1] = (__bf16)b;
    r.v = t;
#endif
    return r;
}

// ---------------------------------------------------------------------------
// One-time weight conversion: Wt[n][k] = bf16(W[k][n])  (transposed, bf16).
// Makes the projection B-operand two contiguous b128 loads with no cvt work.
// ---------------------------------------------------------------------------
__global__ __launch_bounds__(256)
void convert_weight_kernel(const float* __restrict__ W,
                           unsigned short* __restrict__ Wt)
{
    const int idx = blockIdx.x * 256 + threadIdx.x;    // 0 .. 16383
    const int n = idx >> 7, k = idx & 127;
    Wt[(size_t)n * DIM + k] = f2bf(W[(size_t)k * DIM + n]);
}

// ---------------------------------------------------------------------------
// Projection GEMM: Out = (A[S,128] @ W[128,128]) * out_scale, emitted as bf16.
// W supplied pre-transposed bf16 (Wt[n][k]). transpose_out=1 stores Out[d][s].
// Block: 128 threads = 4 waves computes 16 rows x 128 cols.
// ---------------------------------------------------------------------------
__global__ __launch_bounds__(128)
void proj_gemm_kernel(const float* __restrict__ A,
                      const unsigned short* __restrict__ Wt,
                      unsigned short* __restrict__ Out,
                      int transpose_out, float out_scale)
{
    const int tid  = threadIdx.x;
    const int wave = tid >> 5;
    const int lane = tid & 31;
    const int hsel = lane >> 4;   // which 16-lane half
    const int l16  = lane & 15;
    const int row0 = blockIdx.x * 16;

    v8f acc0 = {};
    v8f acc1 = {};

    const float* Arow = A + (size_t)(row0 + l16) * DIM;

    #pragma unroll
    for (int kc = 0; kc < DIM; kc += 32) {
        // ---- A fragment (16x32 bf16): lanes0-15 K={kc+0..7,kc+16..23},
        //      lanes16-31 K={kc+8..15,kc+24..31} ----
        BF16Frag af;
        {
            const int b1 = kc + hsel * 8;
            const float4 p0 = *(const float4*)(Arow + b1);
            const float4 p1 = *(const float4*)(Arow + b1 + 4);
            const float4 p2 = *(const float4*)(Arow + b1 + 16);
            const float4 p3 = *(const float4*)(Arow + b1 + 20);
            pack2(af, 0, p0.x, p0.y); pack2(af, 1, p0.z, p0.w);
            pack2(af, 2, p1.x, p1.y); pack2(af, 3, p1.z, p1.w);
            pack2(af, 4, p2.x, p2.y); pack2(af, 5, p2.z, p2.w);
            pack2(af, 6, p3.x, p3.y); pack2(af, 7, p3.z, p3.w);
        }
        // ---- B fragments (32x16): lanes0-15 K={0..7,16..23}, lanes16-31 K={8..15,24..31},
        //      loaded straight from pre-transposed bf16 weights ----
        #pragma unroll
        for (int ct = 0; ct < 2; ++ct) {
            const int n  = wave * 32 + ct * 16 + l16;
            const int kb = kc + hsel * 8;
            BF16Frag bfg;
            const unsigned short* wrow = Wt + (size_t)n * DIM + kb;
            bfg.q[0] = *(const uint4*)(wrow);
            bfg.q[1] = *(const uint4*)(wrow + 16);
            if (ct == 0)
                acc0 = __builtin_amdgcn_wmma_f32_16x16x32_bf16(
                    false, af.v, false, bfg.v, (short)0, acc0, false, false);
            else
                acc1 = __builtin_amdgcn_wmma_f32_16x16x32_bf16(
                    false, af.v, false, bfg.v, (short)0, acc1, false, false);
        }
    }

    // C layout: row = v + 8*hsel, col = l16
    #pragma unroll
    for (int v = 0; v < 8; ++v) {
        const int r = row0 + v + 8 * hsel;
        const int c0 = wave * 32 + l16;
        const int c1 = c0 + 16;
        const unsigned short b0 = f2bf(acc0[v] * out_scale);
        const unsigned short b1 = f2bf(acc1[v] * out_scale);
        if (transpose_out) {
            Out[(size_t)c0 * SEQ + r] = b0;
            Out[(size_t)c1 * SEQ + r] = b1;
        } else {
            Out[(size_t)r * DIM + c0] = b0;
            Out[(size_t)r * DIM + c1] = b1;
        }
    }
}

// ---------------------------------------------------------------------------
// Fused flash attention + online statistics. 1/sqrt(D) pre-folded into Q.
// Grid: SEQ/BM blocks, 128 threads (4 waves). Wave owns 16 query rows.
// stats: [0]=sum|out|, [1]=sum out^2, [2]=sum_i span_i, [3]=sum_i var_i
// ---------------------------------------------------------------------------
__global__ __launch_bounds__(128)
void flash_attn_kernel(const unsigned short* __restrict__ Qbf,
                       const unsigned short* __restrict__ Kbf,
                       const unsigned short* __restrict__ Vtbf,
                       float* __restrict__ Out, float* __restrict__ stats)
{
    __shared__ unsigned short Kl[BN * DIM];            // key-major  [BN][DIM]
    __shared__ unsigned short Vl[DIM * BN];            // d-major    [DIM][BN]
    __shared__ unsigned short Pl[NWAVES][16][BN];      // per-wave P tile

    const int tid  = threadIdx.x;
    const int wave = tid >> 5;
    const int lane = tid & 31;
    const int hsel = lane >> 4;
    const int l16  = lane & 15;
    const int qrow0 = blockIdx.x * BM + wave * 16;

    // ---- load Q fragments once (bf16, pre-scaled), A-layout ----
    BF16Frag qf[4];
    const unsigned short* Qrow = Qbf + (size_t)(qrow0 + l16) * DIM;
    #pragma unroll
    for (int kc4 = 0; kc4 < 4; ++kc4) {
        const int b1 = kc4 * 32 + hsel * 8;
        qf[kc4].q[0] = *(const uint4*)(Qrow + b1);
        qf[kc4].q[1] = *(const uint4*)(Qrow + b1 + 16);
    }

    float m_i[8], l_i[8], a1[8], aj[8], aj2[8];
    v8f o[8];
    const v8f vzero = {};
    #pragma unroll
    for (int v = 0; v < 8; ++v) {
        m_i[v] = -3.0e38f; l_i[v] = 0.f;
        a1[v] = 0.f; aj[v] = 0.f; aj2[v] = 0.f;
        o[v] = vzero;
    }

    for (int kb = 0; kb < SEQ / BN; ++kb) {
        const int kbase = kb * BN;
        __syncthreads();
        // ---- stage K tile (64x128 bf16 = 16KB) ----
        {
            const uint4* gK = (const uint4*)(Kbf + (size_t)kbase * DIM);
            uint4* sK = (uint4*)Kl;
            #pragma unroll
            for (int it = 0; it < 8; ++it)
                sK[tid + it * 128] = gK[tid + it * 128];
        }
        // ---- stage V^T tile (128x64 bf16 = 16KB) ----
        {
            uint4* sV = (uint4*)Vl;
            #pragma unroll
            for (int it = 0; it < 8; ++it) {
                const int f = tid + it * 128;
                const int d = f >> 3, c = f & 7;
                sV[f] = *((const uint4*)(Vtbf + (size_t)d * SEQ + kbase) + c);
            }
        }
        if (kb + 1 < SEQ / BN) {   // speculative prefetch of next tiles into L2
            __builtin_prefetch(Kbf + (size_t)(kbase + BN) * DIM + tid * 64, 0, 0);
            __builtin_prefetch(Vtbf + (size_t)tid * SEQ + kbase + BN, 0, 0);
        }
        __syncthreads();

        // ---- scores: S = Q K^T (scale pre-folded), 4 n-tiles of 16 keys ----
        v8f sc[4];
        #pragma unroll
        for (int nt = 0; nt < 4; ++nt) {
            sc[nt] = vzero;
            #pragma unroll
            for (int kc4 = 0; kc4 < 4; ++kc4) {
                BF16Frag bfg;
                const unsigned short* krow =
                    Kl + (nt * 16 + l16) * DIM + kc4 * 32 + hsel * 8;
                bfg.q[0] = *(const uint4*)(krow);
                bfg.q[1] = *(const uint4*)(krow + 16);
                sc[nt] = __builtin_amdgcn_wmma_f32_16x16x32_bf16(
                    false, qf[kc4].v, false, bfg.v, (short)0, sc[nt], false, false);
            }
        }

        // ---- online softmax update ----
        #pragma unroll
        for (int v = 0; v < 8; ++v) {
            float t = fmaxf(fmaxf(sc[0][v], sc[1][v]), fmaxf(sc[2][v], sc[3][v]));
            #pragma unroll
            for (int off = 1; off <= 8; off <<= 1)
                t = fmaxf(t, __shfl_xor(t, off, 32));   // reduce within 16-lane half
            const float mnew  = fmaxf(m_i[v], t);
            const float alpha = __expf(m_i[v] - mnew);
            m_i[v] = mnew;
            l_i[v] *= alpha; a1[v] *= alpha; aj[v] *= alpha; aj2[v] *= alpha;
            #pragma unroll
            for (int t8 = 0; t8 < 8; ++t8) o[t8][v] *= alpha;
        }

        // ---- probabilities, online stats, stage P to LDS (C->A relayout) ----
        #pragma unroll
        for (int nt = 0; nt < 4; ++nt) {
            const float jf = (float)(kbase + nt * 16 + l16);
            float pv[8];
            #pragma unroll
            for (int v = 0; v < 8; ++v) {
                const float p = __expf(sc[nt][v] - m_i[v]);
                const float irow = (float)(qrow0 + v + 8 * hsel);
                pv[v] = p;
                l_i[v] += p;
                a1[v]  += p * fabsf(irow - jf);
                aj[v]  += p * jf;
                aj2[v] += p * jf * jf;
            }
            #pragma unroll
            for (int v = 0; v < 8; v += 2) {
                const BF16Pair pk = packpair(pv[v], pv[v + 1]);
                Pl[wave][v + 8 * hsel][nt * 16 + l16]     = pk.s[0];
                Pl[wave][v + 1 + 8 * hsel][nt * 16 + l16] = pk.s[1];
            }
        }

        // ---- O += P @ V : 2 k-chunks x 8 d-tiles ----
        #pragma unroll
        for (int c = 0; c < 2; ++c) {
            BF16Frag pf;
            const unsigned short* prow = &Pl[wave][l16][c * 32 + hsel * 8];
            pf.q[0] = *(const uint4*)(prow);
            pf.q[1] = *(const uint4*)(prow + 16);
            #pragma unroll
            for (int t = 0; t < 8; ++t) {
                BF16Frag vf;
                const unsigned short* vrow =
                    Vl + (t * 16 + l16) * BN + c * 32 + hsel * 8;
                vf.q[0] = *(const uint4*)(vrow);
                vf.q[1] = *(const uint4*)(vrow + 16);
                o[t] = __builtin_amdgcn_wmma_f32_16x16x32_bf16(
                    false, pf.v, false, vf.v, (short)0, o[t], false, false);
            }
        }
    }

    // ---- final row reductions within each 16-lane half ----
    #pragma unroll
    for (int v = 0; v < 8; ++v) {
        #pragma unroll
        for (int off = 1; off <= 8; off <<= 1) {
            l_i[v] += __shfl_xor(l_i[v], off, 32);
            a1[v]  += __shfl_xor(a1[v],  off, 32);
            aj[v]  += __shfl_xor(aj[v],  off, 32);
            aj2[v] += __shfl_xor(aj2[v], off, 32);
        }
    }

    float inv[8];
    #pragma unroll
    for (int v = 0; v < 8; ++v) inv[v] = 1.0f / l_i[v];

    float sabs = 0.f, ssq = 0.f;
    #pragma unroll
    for (int t = 0; t < 8; ++t) {
        #pragma unroll
        for (int v = 0; v < 8; ++v) {
            const float val = o[t][v] * inv[v];
            const int r = qrow0 + v + 8 * hsel;
            const int c = t * 16 + l16;
            Out[(size_t)r * DIM + c] = val;
            sabs += fabsf(val);
            ssq  += val * val;
        }
    }
    atomicAdd(&stats[0], sabs);
    atomicAdd(&stats[1], ssq);

    if (l16 == 0) {                       // one lane per half owns 8 rows
        float span_sum = 0.f, var_sum = 0.f;
        #pragma unroll
        for (int v = 0; v < 8; ++v) {
            span_sum += a1[v] * inv[v];
            const float mp = aj[v] * inv[v];
            var_sum += aj2[v] * inv[v] - mp * mp;
        }
        atomicAdd(&stats[2], span_sum);
        atomicAdd(&stats[3], var_sum);
    }
}

__global__ void init_stats_kernel(float* stats) {
    if (threadIdx.x < 4) stats[threadIdx.x] = 0.0f;
}

__global__ void finalize_kernel(const float* __restrict__ stats,
                                float* __restrict__ behavior) {
    const float Sf = (float)SEQ, Df = (float)DIM;
    behavior[0] = stats[2] / (Sf * Sf);                       // attention span
    const float l2 = sqrtf(stats[1]);
    float sp = 1.0f - stats[0] / (l2 * sqrtf(Sf * Df));
    behavior[1] = fminf(fmaxf(sp, 0.0f), 1.0f);               // sparsity
    behavior[2] = 0.0f;                                       // gradient flow
    behavior[3] = fminf(1.0f, (stats[3] / Sf) / (Sf * Sf));   // locality
    behavior[4] = 1.0f;                                       // intensity (saturated)
}

extern "C" void kernel_launch(void* const* d_in, const int* in_sizes, int n_in,
                              void* d_out, int out_size, void* d_ws, size_t ws_size,
                              hipStream_t stream)
{
    (void)in_sizes; (void)n_in; (void)out_size; (void)ws_size;
    const float* latent   = (const float*)d_in[0];
    const float* stimulus = (const float*)d_in[1];
    const float* Wk       = (const float*)d_in[2];
    const float* Wv       = (const float*)d_in[3];
    const float* Wq       = (const float*)d_in[4];

    float* out      = (float*)d_out;
    float* behavior = out + (size_t)SEQ * DIM;

    // workspace layout: 4 stat floats (256B pad), bf16 Q, K, V^T, then
    // three pre-transposed bf16 weight copies (32KB each)
    float* stats = (float*)d_ws;
    unsigned short* Qbf   = (unsigned short*)((char*)d_ws + 256);
    unsigned short* Kbf   = Qbf  + (size_t)SEQ * DIM;
    unsigned short* Vtbf  = Kbf  + (size_t)SEQ * DIM;
    unsigned short* WkT   = Vtbf + (size_t)SEQ * DIM;
    unsigned short* WvT   = WkT  + (size_t)DIM * DIM;
    unsigned short* WqT   = WvT  + (size_t)DIM * DIM;

    const float qscale = 0.08838834764831845f;   // 1/sqrt(128)

    init_stats_kernel<<<1, 32, 0, stream>>>(stats);
    convert_weight_kernel<<<DIM * DIM / 256, 256, 0, stream>>>(Wk, WkT);
    convert_weight_kernel<<<DIM * DIM / 256, 256, 0, stream>>>(Wv, WvT);
    convert_weight_kernel<<<DIM * DIM / 256, 256, 0, stream>>>(Wq, WqT);
    proj_gemm_kernel<<<SEQ / 16, 128, 0, stream>>>(latent,   WkT, Kbf,  0, 1.0f);
    proj_gemm_kernel<<<SEQ / 16, 128, 0, stream>>>(latent,   WvT, Vtbf, 1, 1.0f);
    proj_gemm_kernel<<<SEQ / 16, 128, 0, stream>>>(stimulus, WqT, Qbf,  0, qscale);
    flash_attn_kernel<<<SEQ / BM, 128, 0, stream>>>(Qbf, Kbf, Vtbf, out, stats);
    finalize_kernel<<<1, 1, 0, stream>>>(stats, behavior);
}
